// OldModel_39479339385284
// MI455X (gfx1250) — matrix operand order repaired and checked by reading, not compiled, not run on hardware
//
#include <hip/hip_runtime.h>
#include <hip/hip_bf16.h>

#define VOCAB 32000
#define DIM   512
#define NH    4
#define NB    4
#define SEQ   512
#define NLAYER 4

// fused vocab-attention blocking
#define MT      64                     // rows per workgroup (4 row-tiles)
#define VSPLIT  10                     // vocab split (flash split-K)
#define VCHUNK  128                    // vocab columns per iteration
#define NCHUNKS (VOCAB / VCHUNK)       // 250
#define CPS     (NCHUNKS / VSPLIT)     // 25 chunks per split
#define FK_LD   520                    // 512 + 8 pad (16B aligned, kills bank alias)
#define P_LD    136                    // 128 + 8 pad

#if __has_builtin(__builtin_amdgcn_global_load_async_to_lds_b128)
#define HAS_ASYNC_LDS 1
#else
#define HAS_ASYNC_LDS 0
#endif

typedef __bf16 bf16_t;
typedef __attribute__((ext_vector_type(16))) __bf16 v16bf;
typedef __attribute__((ext_vector_type(8)))  __bf16 v8bf;
typedef __attribute__((ext_vector_type(8)))  float  v8f;
typedef int v4i_t __attribute__((ext_vector_type(4)));

// ---------------------------------------------------------------------------
// WMMA fragment helpers (wave32 layouts per CDNA5 ISA 7.12.2)
// ---------------------------------------------------------------------------
__device__ __forceinline__ v16bf load_frag_a(const bf16_t* base, int ld, int lane) {
  const bf16_t* p = base + (long)(lane & 15) * ld + ((lane >> 4) << 3);
  v8bf lo = *(const v8bf*)(p);
  v8bf hi = *(const v8bf*)(p + 16);
  v16bf r;
#pragma unroll
  for (int i = 0; i < 8; ++i) { r[i] = lo[i]; r[i + 8] = hi[i]; }
  return r;
}

__device__ __forceinline__ v16bf load_frag_b(const bf16_t* baseT, int ld, int lane) {
  const bf16_t* p = baseT + (long)(lane & 15) * ld + ((lane >> 4) << 3);
  v8bf lo = *(const v8bf*)(p);
  v8bf hi = *(const v8bf*)(p + 16);
  v16bf r;
#pragma unroll
  for (int i = 0; i < 8; ++i) { r[i] = lo[i]; r[i + 8] = hi[i]; }
  return r;
}

__device__ __forceinline__ v8f wmma_bf16(v16bf a, v16bf b, v8f c) {
  return __builtin_amdgcn_wmma_f32_16x16x32_bf16(false, a, false, b, (short)0, c,
                                                 false, false);
}

// ---------------------------------------------------------------------------
// Conversion / transpose kernels
// ---------------------------------------------------------------------------
__global__ void k_cvt_wte(const float* __restrict__ wte, bf16_t* __restrict__ w,
                          bf16_t* __restrict__ wT) {
  long i = (long)blockIdx.x * 256 + threadIdx.x;
  if (i >= (long)VOCAB * DIM) return;
  int v = (int)(i / DIM), d = (int)(i % DIM);
  bf16_t b = (bf16_t)wte[i];
  w[i] = b;
  wT[(long)d * VOCAB + v] = b;
}

__global__ void k_cvt_wh(const float* __restrict__ W, bf16_t* __restrict__ WT) {
  long i = (long)blockIdx.x * 256 + threadIdx.x;
  if (i >= (long)NH * DIM * DIM) return;
  int h = (int)(i / (DIM * DIM));
  int d = (int)((i / DIM) % DIM);
  int e = (int)(i % DIM);
  WT[((long)h * DIM + e) * DIM + d] = (bf16_t)W[i];
}

__global__ void k_cvt_wo(const float* __restrict__ W, bf16_t* __restrict__ Wb) {
  long i = (long)blockIdx.x * 256 + threadIdx.x;
  if (i >= (long)DIM * NH * DIM) return;
  Wb[i] = (bf16_t)W[i];
}

// ---------------------------------------------------------------------------
// LayerNorm kernels
// ---------------------------------------------------------------------------
__device__ __forceinline__ float block_reduce_sum256(float v, float* red) {
  int t = threadIdx.x;
  red[t] = v; __syncthreads();
#pragma unroll
  for (int off = 128; off > 0; off >>= 1) {
    if (t < off) red[t] += red[t + off];
    __syncthreads();
  }
  float r = red[0];
  __syncthreads();
  return r;
}

__global__ void k_embed_ln(const int* __restrict__ x, const float* __restrict__ wte,
                           const float* __restrict__ g, float* __restrict__ e) {
  __shared__ float red[256];
  int row = blockIdx.x;
  int tok = x[row];
  const float* src = wte + (long)tok * DIM;
  int t = threadIdx.x;
  float v0 = src[t], v1 = src[t + 256];
  float mean = block_reduce_sum256(v0 + v1, red) * (1.0f / DIM);
  float d0 = v0 - mean, d1 = v1 - mean;
  float var = block_reduce_sum256(d0 * d0 + d1 * d1, red) * (1.0f / DIM);
  float rstd = rsqrtf(var + 1e-5f);
  float* dst = e + (long)row * DIM;
  dst[t]       = d0 * rstd * g[t];
  dst[t + 256] = d1 * rstd * g[t + 256];
}

__global__ void k_pos_ln(const float* __restrict__ wpe, const float* __restrict__ g,
                         bf16_t* __restrict__ xi, bf16_t* __restrict__ xj) {
  __shared__ float red[256];
  int row = blockIdx.x;
  const float* src = wpe + (long)row * DIM;
  int t = threadIdx.x;
  float v0 = src[t], v1 = src[t + 256];
  float mean = block_reduce_sum256(v0 + v1, red) * (1.0f / DIM);
  float d0 = v0 - mean, d1 = v1 - mean;
  float var = block_reduce_sum256(d0 * d0 + d1 * d1, red) * (1.0f / DIM);
  float rstd = rsqrtf(var + 1e-5f);
  float o0 = d0 * rstd * g[t], o1 = d1 * rstd * g[t + 256];
  if (row < SEQ) {
    xi[(long)row * DIM + t]       = (bf16_t)o0;
    xi[(long)row * DIM + t + 256] = (bf16_t)o1;
  }
  if (row >= 1) {
    xj[(long)(row - 1) * DIM + t]       = (bf16_t)o0;
    xj[(long)(row - 1) * DIM + t + 256] = (bf16_t)o1;
  }
}

// ---------------------------------------------------------------------------
// Q/K projections
// ---------------------------------------------------------------------------
__global__ void k_qk(const bf16_t* __restrict__ xj, const bf16_t* __restrict__ xi,
                     const bf16_t* __restrict__ WqT, const bf16_t* __restrict__ WkT,
                     bf16_t* __restrict__ Q, bf16_t* __restrict__ K) {
  int slot = blockIdx.y;
  bool isK = slot >= NH;
  int h = slot & (NH - 1);
  const bf16_t* A  = isK ? xi : xj;
  const bf16_t* BT = (isK ? WkT : WqT) + (long)h * DIM * DIM;
  bf16_t* C = (isK ? K : Q) + (long)h * SEQ * DIM;
  int tile = blockIdx.x;
  int m0 = (tile / (DIM / 64)) * 16;
  int n0 = (tile % (DIM / 64)) * 64 + (threadIdx.x >> 5) * 16;
  int lane = threadIdx.x & 31;
  v8f acc = {};
  for (int k0 = 0; k0 < DIM; k0 += 32) {
    v16bf a = load_frag_a(A + (long)m0 * DIM + k0, DIM, lane);
    v16bf b = load_frag_b(BT + (long)n0 * DIM + k0, DIM, lane);
    acc = wmma_bf16(a, b, acc);
  }
  int col = n0 + (lane & 15);
  int rb  = m0 + ((lane >> 4) << 3);
#pragma unroll
  for (int r = 0; r < 8; ++r) C[(long)(rb + r) * DIM + col] = (bf16_t)acc[r];
}

// ---------------------------------------------------------------------------
// krn = softmax(mask(clip(Q K^T / sqrt(D))))
// ---------------------------------------------------------------------------
__global__ void k_krn(const bf16_t* __restrict__ Q, const bf16_t* __restrict__ K,
                      bf16_t* __restrict__ krn) {
  __shared__ float sbuf[16][SEQ];
  int s0 = blockIdx.x * 16;
  int h  = blockIdx.y;
  int w = threadIdx.x >> 5, lane = threadIdx.x & 31;
  int tlim = s0 + 16;
  const bf16_t* Qh = Q + (long)h * SEQ * DIM;
  const bf16_t* Kh = K + (long)h * SEQ * DIM;
  const float scale = 0.04419417382415922f;
  for (int t0 = w * 16; t0 < tlim; t0 += 64) {
    v8f acc = {};
    for (int k0 = 0; k0 < DIM; k0 += 32) {
      v16bf a = load_frag_a(Qh + (long)s0 * DIM + k0, DIM, lane);
      v16bf b = load_frag_b(Kh + (long)t0 * DIM + k0, DIM, lane);
      acc = wmma_bf16(a, b, acc);
    }
    int col = t0 + (lane & 15);
    int rb = (lane >> 4) << 3;
#pragma unroll
    for (int r = 0; r < 8; ++r) {
      int row = rb + r;
      float v = acc[r] * scale;
      v = fminf(10.0f, fmaxf(-10.0f, v));
      if (col > s0 + row) v = -1e30f;
      sbuf[row][col] = v;
    }
  }
  __syncthreads();
  for (int row = w * 4; row < w * 4 + 4; ++row) {
    int s = s0 + row;
    float m = -1e30f;
    for (int c = lane; c < tlim; c += 32) m = fmaxf(m, sbuf[row][c]);
#pragma unroll
    for (int off = 16; off; off >>= 1) m = fmaxf(m, __shfl_xor(m, off, 32));
    float l = 0.f;
    for (int c = lane; c < tlim; c += 32) {
      float p = __expf(sbuf[row][c] - m);
      sbuf[row][c] = p;
      l += p;
    }
#pragma unroll
    for (int off = 16; off; off >>= 1) l += __shfl_xor(l, off, 32);
    float inv = 1.0f / l;
    for (int c = lane; c < tlim; c += 32)
      krn[((long)h * SEQ + s) * SEQ + c] = (bf16_t)(sbuf[row][c] * inv);
  }
}

// ---------------------------------------------------------------------------
// Fused vocab attention, M-blocked + vocab-split flash.
// f_k tile staged to LDS via GLOBAL_LOAD_ASYNC_TO_LDS_B128 when available
// (ASYNCcnt path; DMA overlapped with accumulator init, closed by
// s_wait_asynccnt 0 before the barrier). Fallback: sync copy.
// ---------------------------------------------------------------------------
__global__ void __launch_bounds__(256, 1)
k_vattn(const bf16_t* __restrict__ fkb, const bf16_t* __restrict__ wteb,
        const bf16_t* __restrict__ wteTb, float* __restrict__ expart,
        float* __restrict__ mlpart) {
  __shared__ bf16_t fkt[MT * FK_LD];     // 66.5 KB
  __shared__ bf16_t Pbuf[MT * P_LD];     // 17.4 KB
  __shared__ float  wmax[8][MT];
  __shared__ float  m_s[MT], scale_s[MT];
  __shared__ float  lsum[8][MT];

  int sp    = blockIdx.x % VSPLIT;
  int rbidx = blockIdx.x / VSPLIT;
  int b  = rbidx >> 3;
  int s0 = (rbidx & 7) * MT;
  int tid = threadIdx.x;
  int w = tid >> 5, lane = tid & 31;
  int rb = (lane >> 4) << 3;
  int dbase = w * 64;                    // this wave's D slice

  // stage f_k tile (MT x DIM) into LDS
  const bf16_t* src = fkb + ((long)b * SEQ + s0) * DIM;
#if HAS_ASYNC_LDS
  for (int c = tid; c < MT * (DIM / 8); c += 256) {
    int row = c >> 6;                    // 64 chunks of 8 elems per row
    int ch  = c & 63;
    __builtin_amdgcn_global_load_async_to_lds_b128(
        (__attribute__((address_space(1))) v4i_t*)(src + (long)row * DIM + ch * 8),
        (__attribute__((address_space(3))) v4i_t*)(&fkt[row * FK_LD + ch * 8]),
        /*offset=*/0, /*cpol=*/0);
  }
#else
  for (int c = tid; c < MT * (DIM / 8); c += 256) {
    int row = c >> 6;
    int ch  = c & 63;
    *(v8bf*)(&fkt[row * FK_LD + ch * 8]) =
        *(const v8bf*)(src + (long)row * DIM + ch * 8);
  }
#endif
  if (tid < MT) m_s[tid] = -1e30f;

  v8f exacc[4][4];
#pragma unroll
  for (int rt = 0; rt < 4; ++rt)
#pragma unroll
    for (int dj = 0; dj < 4; ++dj) exacc[rt][dj] = (v8f){};
  float lacc[4][8];
#pragma unroll
  for (int rt = 0; rt < 4; ++rt)
#pragma unroll
    for (int r = 0; r < 8; ++r) lacc[rt][r] = 0.f;

#if HAS_ASYNC_LDS
#if __has_builtin(__builtin_amdgcn_s_wait_asynccnt)
  __builtin_amdgcn_s_wait_asynccnt(0);
#else
  asm volatile("s_wait_asynccnt 0" ::: "memory");
#endif
#endif
  __syncthreads();

  for (int it = 0; it < CPS; ++it) {
    int v0 = (sp * CPS + it) * VCHUNK;
    int vb = v0 + w * 16;
    // ---- scores: 4 row-tiles x this wave's 16 vocab cols ----
    v8f sacc[4];
#pragma unroll
    for (int rt = 0; rt < 4; ++rt) sacc[rt] = (v8f){};
    for (int k = 0; k < 16; ++k) {
      v16bf bfr = load_frag_b(wteb + (long)vb * DIM + k * 32, DIM, lane);
#pragma unroll
      for (int rt = 0; rt < 4; ++rt) {
        v16bf af = load_frag_a(&fkt[(rt * 16) * FK_LD + k * 32], FK_LD, lane);
        sacc[rt] = wmma_bf16(af, bfr, sacc[rt]);
      }
    }
    // ---- per-row partial max -> LDS ----
#pragma unroll
    for (int rt = 0; rt < 4; ++rt)
#pragma unroll
      for (int r = 0; r < 8; ++r) {
        float m = sacc[rt][r];
        m = fmaxf(m, __shfl_xor(m, 1, 32));
        m = fmaxf(m, __shfl_xor(m, 2, 32));
        m = fmaxf(m, __shfl_xor(m, 4, 32));
        m = fmaxf(m, __shfl_xor(m, 8, 32));
        if ((lane & 15) == 0) wmax[w][rt * 16 + rb + r] = m;
      }
    __syncthreads();
    if (tid < MT) {
      float mo = m_s[tid];
      float mn = mo;
#pragma unroll
      for (int i = 0; i < 8; ++i) mn = fmaxf(mn, wmax[i][tid]);
      m_s[tid] = mn;
      scale_s[tid] = __expf(mo - mn);
    }
    __syncthreads();
    // ---- P = exp(s - m); rescale accumulators; stage P ----
#pragma unroll
    for (int rt = 0; rt < 4; ++rt)
#pragma unroll
      for (int r = 0; r < 8; ++r) {
        int row = rt * 16 + rb + r;
        float mn = m_s[row], sc = scale_s[row];
        float p = __expf(sacc[rt][r] - mn);
#pragma unroll
        for (int dj = 0; dj < 4; ++dj) exacc[rt][dj][r] *= sc;
        float ps = p;
        ps += __shfl_xor(ps, 1, 32);
        ps += __shfl_xor(ps, 2, 32);
        ps += __shfl_xor(ps, 4, 32);
        ps += __shfl_xor(ps, 8, 32);
        lacc[rt][r] = lacc[rt][r] * sc + ps;
        Pbuf[row * P_LD + w * 16 + (lane & 15)] = (bf16_t)p;
      }
    __syncthreads();
    // ---- ex += P(64x128) . wte(128 x Dslice)  via wteT ----
#pragma unroll
    for (int kk = 0; kk < 4; ++kk) {
      v16bf pa[4];
#pragma unroll
      for (int rt = 0; rt < 4; ++rt)
        pa[rt] = load_frag_a(&Pbuf[(rt * 16) * P_LD + kk * 32], P_LD, lane);
#pragma unroll
      for (int dj = 0; dj < 4; ++dj) {
        v16bf bfr = load_frag_b(
            wteTb + (long)(dbase + dj * 16) * VOCAB + v0 + kk * 32, VOCAB, lane);
#pragma unroll
        for (int rt = 0; rt < 4; ++rt)
          exacc[rt][dj] = wmma_bf16(pa[rt], bfr, exacc[rt][dj]);
      }
    }
    __syncthreads();
  }
  // ---- emit split partials ----
  if ((lane & 15) == 0) {
#pragma unroll
    for (int rt = 0; rt < 4; ++rt)
#pragma unroll
      for (int r = 0; r < 8; ++r) lsum[w][rt * 16 + rb + r] = lacc[rt][r];
  }
  __syncthreads();
  if (tid < MT) {
    float l = 0.f;
#pragma unroll
    for (int i = 0; i < 8; ++i) l += lsum[i][tid];
    long gr = (long)b * SEQ + s0 + tid;
    mlpart[((long)sp * (NB * SEQ) + gr) * 2 + 0] = m_s[tid];
    mlpart[((long)sp * (NB * SEQ) + gr) * 2 + 1] = l;
  }
#pragma unroll
  for (int rt = 0; rt < 4; ++rt)
#pragma unroll
    for (int r = 0; r < 8; ++r) {
      long gr = (long)b * SEQ + s0 + rt * 16 + rb + r;
#pragma unroll
      for (int dj = 0; dj < 4; ++dj) {
        int d = dbase + dj * 16 + (lane & 15);
        expart[((long)sp * (NB * SEQ) + gr) * DIM + d] = exacc[rt][dj][r];
      }
    }
}

// ---------------------------------------------------------------------------
// Combine vocab-split partials; write VmT[b,d,s] = e - ex
// ---------------------------------------------------------------------------
__global__ void k_vcombine(const float* __restrict__ expart,
                           const float* __restrict__ mlpart,
                           const float* __restrict__ e, bf16_t* __restrict__ VmT) {
  __shared__ float wsc[VSPLIT];
  int gr = blockIdx.x;                   // 0..NB*SEQ-1
  int b = gr / SEQ, s = gr % SEQ;
  if (threadIdx.x == 0) {
    float mv[VSPLIT], lv[VSPLIT];
    float M = -1e30f;
#pragma unroll
    for (int i = 0; i < VSPLIT; ++i) {
      mv[i] = mlpart[((long)i * (NB * SEQ) + gr) * 2 + 0];
      lv[i] = mlpart[((long)i * (NB * SEQ) + gr) * 2 + 1];
      M = fmaxf(M, mv[i]);
    }
    float L = 0.f;
    float wv[VSPLIT];
#pragma unroll
    for (int i = 0; i < VSPLIT; ++i) {
      wv[i] = __expf(mv[i] - M);
      L += wv[i] * lv[i];
    }
    float invL = 1.0f / L;
#pragma unroll
    for (int i = 0; i < VSPLIT; ++i) wsc[i] = wv[i] * invL;
  }
  __syncthreads();
  for (int d = threadIdx.x; d < DIM; d += blockDim.x) {
    float ex = 0.f;
#pragma unroll
    for (int i = 0; i < VSPLIT; ++i)
      ex += wsc[i] * expart[((long)i * (NB * SEQ) + gr) * DIM + d];
    float vm = e[(long)gr * DIM + d] - ex;
    VmT[((long)b * DIM + d) * SEQ + s] = (bf16_t)vm;
  }
}

// ---------------------------------------------------------------------------
// U[b,s,h*D+d] = coef[s] * sum_t krn[h,s,t] * Vm[b,t,d]
// ---------------------------------------------------------------------------
__global__ void k_delta(const bf16_t* __restrict__ krn, const bf16_t* __restrict__ VmT,
                        bf16_t* __restrict__ U) {
  int s0 = blockIdx.x * 16;
  int h = blockIdx.y, b = blockIdx.z;
  int w = threadIdx.x >> 5, lane = threadIdx.x & 31;
  int nt = (s0 + 16 + 31) & ~31;
  const bf16_t* A  = krn + ((long)h * SEQ + s0) * SEQ;
  const bf16_t* BT = VmT + (long)b * DIM * SEQ;
  v8f acc[8];
#pragma unroll
  for (int j = 0; j < 8; ++j) acc[j] = (v8f){};
  for (int t0 = 0; t0 < nt; t0 += 32) {
    v16bf a = load_frag_a(A + t0, SEQ, lane);
#pragma unroll
    for (int j = 0; j < 8; ++j) {
      v16bf bfr = load_frag_b(BT + (long)(w * 128 + j * 16) * SEQ + t0, SEQ, lane);
      acc[j] = wmma_bf16(a, bfr, acc[j]);
    }
  }
  int rb = (lane >> 4) << 3;
#pragma unroll
  for (int r = 0; r < 8; ++r) {
    int s = s0 + rb + r;
    float coef = 1.0f / (1.0f + (float)s);
#pragma unroll
    for (int j = 0; j < 8; ++j) {
      int d = w * 128 + j * 16 + (lane & 15);
      U[((long)b * SEQ + s) * (NH * DIM) + (long)h * DIM + d] =
          (bf16_t)(acc[j][r] * coef);
    }
  }
}

// ---------------------------------------------------------------------------
// f_k += U . W_o^T
// ---------------------------------------------------------------------------
__global__ void k_fkupd(const bf16_t* __restrict__ U, const bf16_t* __restrict__ Wo,
                        float* __restrict__ fk, bf16_t* __restrict__ fkb) {
  int s0 = (blockIdx.x >> 3) * 16;
  int n0 = (blockIdx.x & 7) * 64 + (threadIdx.x >> 5) * 16;
  int b = blockIdx.y;
  int lane = threadIdx.x & 31;
  const bf16_t* A = U + ((long)b * SEQ + s0) * (NH * DIM);
  v8f acc = {};
  for (int k0 = 0; k0 < NH * DIM; k0 += 32) {
    v16bf a   = load_frag_a(A + k0, NH * DIM, lane);
    v16bf bfr = load_frag_b(Wo + (long)n0 * (NH * DIM) + k0, NH * DIM, lane);
    acc = wmma_bf16(a, bfr, acc);
  }
  int col = n0 + (lane & 15);
  int rb = (lane >> 4) << 3;
#pragma unroll
  for (int r = 0; r < 8; ++r) {
    long off = ((long)b * SEQ + s0 + rb + r) * DIM + col;
    float nv = fk[off] + acc[r];
    fk[off] = nv;
    fkb[off] = (bf16_t)nv;
  }
}

// ---------------------------------------------------------------------------
// Final LN (last position) + logits
// ---------------------------------------------------------------------------
__global__ void k_final_ln(const float* __restrict__ fk, const float* __restrict__ g,
                           float* __restrict__ lnrow) {
  __shared__ float red[256];
  int b = blockIdx.x;
  const float* src = fk + ((long)b * SEQ + (SEQ - 1)) * DIM;
  int t = threadIdx.x;
  float v0 = src[t], v1 = src[t + 256];
  float mean = block_reduce_sum256(v0 + v1, red) * (1.0f / DIM);
  float d0 = v0 - mean, d1 = v1 - mean;
  float var = block_reduce_sum256(d0 * d0 + d1 * d1, red) * (1.0f / DIM);
  float rstd = rsqrtf(var + 1e-5f);
  lnrow[(long)b * DIM + t]       = d0 * rstd * g[t];
  lnrow[(long)b * DIM + t + 256] = d1 * rstd * g[t + 256];
}

__global__ void k_logits(const float* __restrict__ lnrow, const float* __restrict__ wte,
                         float* __restrict__ out) {
  __shared__ float q[DIM];
  int b = blockIdx.y;
  q[threadIdx.x]       = lnrow[(long)b * DIM + threadIdx.x];
  q[threadIdx.x + 256] = lnrow[(long)b * DIM + threadIdx.x + 256];
  __syncthreads();
  int v = blockIdx.x * 256 + threadIdx.x;
  if (v >= VOCAB) return;
  const float* wr = wte + (long)v * DIM;
  float acc = 0.f;
#pragma unroll 4
  for (int d = 0; d < DIM; ++d) acc += q[d] * wr[d];
  out[(long)b * VOCAB + v] = acc;
}

// ---------------------------------------------------------------------------
extern "C" void kernel_launch(void* const* d_in, const int* in_sizes, int n_in,
                              void* d_out, int out_size, void* d_ws, size_t ws_size,
                              hipStream_t stream) {
  (void)in_sizes; (void)n_in; (void)out_size; (void)ws_size;
  const int*   x    = (const int*)d_in[0];
  const float* wte  = (const float*)d_in[1];
  const float* wpe  = (const float*)d_in[2];
  const float* g_e  = (const float*)d_in[3];
  const float* g_p  = (const float*)d_in[4];
  const float* g_f  = (const float*)d_in[5];
  const float* W_q  = (const float*)d_in[6];
  const float* W_k  = (const float*)d_in[7];
  const float* W_o  = (const float*)d_in[8];
  float* out = (float*)d_out;

  size_t off = 0;
  auto take = [&](size_t bytes) -> void* {
    void* p = (char*)d_ws + off;
    off += (bytes + 255) & ~(size_t)255;
    return p;
  };
  bf16_t* wte_b  = (bf16_t*)take((size_t)VOCAB * DIM * 2);
  bf16_t* wteT_b = (bf16_t*)take((size_t)VOCAB * DIM * 2);
  float*  e_f    = (float*) take((size_t)NB * SEQ * DIM * 4);
  bf16_t* xi_b   = (bf16_t*)take((size_t)SEQ * DIM * 2);
  bf16_t* xj_b   = (bf16_t*)take((size_t)SEQ * DIM * 2);
  bf16_t* WqT_b  = (bf16_t*)take((size_t)NH * DIM * DIM * 2);
  bf16_t* WkT_b  = (bf16_t*)take((size_t)NH * DIM * DIM * 2);
  bf16_t* Wo_b   = (bf16_t*)take((size_t)DIM * NH * DIM * 2);
  bf16_t* Q_b    = (bf16_t*)take((size_t)NH * SEQ * DIM * 2);
  bf16_t* K_b    = (bf16_t*)take((size_t)NH * SEQ * DIM * 2);
  bf16_t* krn_b  = (bf16_t*)take((size_t)NH * SEQ * SEQ * 2);
  float*  fk_f   = (float*) take((size_t)NB * SEQ * DIM * 4);
  bf16_t* fk_b   = (bf16_t*)take((size_t)NB * SEQ * DIM * 2);
  bf16_t* VmT_b  = (bf16_t*)take((size_t)NB * DIM * SEQ * 2);
  bf16_t* U_b    = (bf16_t*)take((size_t)NB * SEQ * NH * DIM * 2);
  float*  lnrow  = (float*) take((size_t)NB * DIM * 4);
  float*  expart = (float*) take((size_t)VSPLIT * NB * SEQ * DIM * 4);  // 42 MB
  float*  mlpart = (float*) take((size_t)VSPLIT * NB * SEQ * 2 * 4);

  (void)hipMemsetAsync(fk_f,  0, (size_t)NB * SEQ * DIM * 4, stream);
  (void)hipMemsetAsync(fk_b,  0, (size_t)NB * SEQ * DIM * 2, stream);
  (void)hipMemsetAsync(krn_b, 0, (size_t)NH * SEQ * SEQ * 2, stream);

  k_cvt_wte<<<(VOCAB * DIM + 255) / 256, 256, 0, stream>>>(wte, wte_b, wteT_b);
  k_cvt_wh <<<(NH * DIM * DIM + 255) / 256, 256, 0, stream>>>(W_q, WqT_b);
  k_cvt_wh <<<(NH * DIM * DIM + 255) / 256, 256, 0, stream>>>(W_k, WkT_b);
  k_cvt_wo <<<(DIM * NH * DIM + 255) / 256, 256, 0, stream>>>(W_o, Wo_b);

  k_embed_ln<<<NB * SEQ, 256, 0, stream>>>(x, wte, g_e, e_f);
  k_pos_ln  <<<SEQ + 1, 256, 0, stream>>>(wpe, g_p, xi_b, xj_b);

  k_qk <<<dim3((SEQ / 16) * (DIM / 64), 2 * NH), 128, 0, stream>>>(
      xj_b, xi_b, WqT_b, WkT_b, Q_b, K_b);
  k_krn<<<dim3(SEQ / 16, NH), 128, 0, stream>>>(Q_b, K_b, krn_b);

  for (int layer = 0; layer < NLAYER; ++layer) {
    k_vattn<<<NB * (SEQ / MT) * VSPLIT, 256, 0, stream>>>(
        fk_b, wte_b, wteT_b, expart, mlpart);
    k_vcombine<<<NB * SEQ, 128, 0, stream>>>(expart, mlpart, e_f, VmT_b);
    k_delta<<<dim3(SEQ / 16, NH, NB), 128, 0, stream>>>(krn_b, VmT_b, U_b);
    k_fkupd<<<dim3((SEQ / 16) * (DIM / 64), NB), 128, 0, stream>>>(
        U_b, Wo_b, fk_f, fk_b);
  }

  k_final_ln<<<NB, 256, 0, stream>>>(fk_f, g_f, lnrow);
  k_logits<<<dim3((VOCAB + 255) / 256, NB), 256, 0, stream>>>(lnrow, wte, out);
}
